// RouteNet_Fermi_wavelet_multiple_level_34462817583673
// MI455X (gfx1250) — matrix operand order repaired and checked by reading, not compiled, not run on hardware
//
#include <hip/hip_runtime.h>

// ---------------- problem constants ----------------
#define P_N 40000
#define T_N 8
#define S_N 32
#define L_N 6000
#define Q_N 6000
#define K_N 56
#define D_N 32

typedef __attribute__((ext_vector_type(16))) _Float16 v16h;
typedef __attribute__((ext_vector_type(8)))  _Float16 v8h;
typedef __attribute__((ext_vector_type(8)))  float    v8f;

// ---------------- workspace layout ----------------
// fp16 region (element offsets in halves)
static constexpr size_t HV_WT0RK = 0;                     // 32x96 rk -> 6 frags *512
static constexpr size_t HV_WT1RK = HV_WT0RK + 3072;
static constexpr size_t HV_PGK   = HV_WT1RK + 3072;       // 64x96 -> 12 frags
static constexpr size_t HV_PGRK  = HV_PGK   + 6144;
static constexpr size_t HV_QGK   = HV_PGRK  + 3072;
static constexpr size_t HV_QGRK  = HV_QGK   + 3072;
static constexpr size_t HV_LGK   = HV_QGRK  + 3072;
static constexpr size_t HV_LGRK  = HV_LGK   + 3072;
static constexpr size_t HV_PEW   = HV_LGRK  + 3072;       // 64x32 -> 4 frags
static constexpr size_t HV_RO1   = HV_PEW   + 2048;       // 32x16 -> 1 frag
static constexpr size_t HV_RO2   = HV_RO1   + 512;        // 16x16 zero-padded to 32x16
static constexpr size_t HV_ENC   = HV_RO2   + 512;        // P x 64 f16
static constexpr size_t HV_QF16  = HV_ENC   + (size_t)P_N * 64;
static constexpr size_t HV_LF16  = HV_QF16  + (size_t)Q_N * 32;
static constexpr size_t HV_END   = HV_LF16  + (size_t)L_N * 32;
static constexpr size_t F32_BASE_BYTES = ((HV_END * 2 + 255) / 256) * 256;
// fp32 region (element offsets in floats)
static constexpr size_t FO_PS    = 0;                          // path_state P x 32
static constexpr size_t FO_PSS   = FO_PS  + (size_t)P_N * 32;  // P x 9 x 32
static constexpr size_t FO_QS    = FO_PSS + (size_t)P_N * 9 * 32;
static constexpr size_t FO_LS    = FO_QS  + (size_t)Q_N * 32;
static constexpr size_t FO_LOAD  = FO_LS  + (size_t)L_N * 32;

// ---------------- helpers ----------------
__device__ __forceinline__ float fast_rcp(float x) { return __builtin_amdgcn_rcpf(x); }
__device__ __forceinline__ float sigm(float x) { return fast_rcp(1.0f + __expf(-x)); }
__device__ __forceinline__ float tanh_f(float x) {
  x = fminf(fmaxf(x, -15.0f), 15.0f);
  float e = __expf(2.0f * x);
  return (e - 1.0f) * fast_rcp(e + 1.0f);
}
__device__ __forceinline__ v8f splat8(float x) {
  v8f c;
#pragma unroll
  for (int i = 0; i < 8; ++i) c[i] = x;
  return c;
}
// A-fragment (16x32 f16) from a contiguous 32-half row starting at `row`
// base = 0 for lanes 0..15, 8 for lanes 16..31
__device__ __forceinline__ v16h make_a(const _Float16* row, int base) {
  v8h lo = *(const v8h*)(row + base);
  v8h hh = *(const v8h*)(row + base + 16);
  v16h a;
#pragma unroll
  for (int i = 0; i < 8; ++i) { a[i] = lo[i]; a[i + 8] = hh[i]; }
  return a;
}
__device__ __forceinline__ v16h load_bfrag(const _Float16* pack, int frag) {
  int lane = threadIdx.x & 31;
  return *(const v16h*)(pack + ((size_t)frag * 32 + lane) * 16);
}
__device__ __forceinline__ v8f wmma16(v16h a, v16h b, v8f c) {
  return __builtin_amdgcn_wmma_f32_16x16x32_f16(false, a, false, b, (short)0, c, false, false);
}
#define WAVE_LDS_FENCE() asm volatile("s_wait_dscnt 0" ::: "memory")

// ---------------- weight packing ----------------
// pack W (K x N, f32 row-major) into B fragments (frag = nt*KT + kt)
// pack[((frag*32+lane)*16 + half)] = W[kt*32 + 8*(lane>>4) + (half&7) + 16*(half>>3)][nt*16 + (lane&15)]
__device__ void pack_mat(const float* W, _Float16* dst, int K, int N) {
  int KT = K / 32, NT = N / 16;
  int total = KT * NT * 512;
  for (int i = threadIdx.x; i < total; i += blockDim.x) {
    int half = i & 15;
    int lane = (i >> 4) & 31;
    int f    = i >> 9;
    int kt = f % KT, nt = f / KT;
    int v = half >> 1, s = half & 1;
    int k = kt * 32 + 8 * (lane >> 4) + (half & 7) + 16 * (half >> 3);
    int n = nt * 16 + (lane & 15);
    dst[i] = (_Float16)W[(size_t)k * N + n];
  }
}
// pack W (16 x 16 row-major), zero-padded in K to 32x16 (single fragment)
__device__ void pack_mat_pad16(const float* W, _Float16* dst) {
  for (int i = threadIdx.x; i < 512; i += blockDim.x) {
    int half = i & 15;
    int lane = (i >> 4) & 31;
    int v = half >> 1, s = half & 1;
    int k = 8 * (lane >> 4) + (half & 7) + 16 * (half >> 3);
    int n = lane & 15;
    dst[i] = (k < 16) ? (_Float16)W[(size_t)k * 16 + n] : (_Float16)0.f;
  }
}

__global__ void pack_kernel(const float* wt0_rk, const float* wt1_rk,
                            const float* pg_k, const float* pg_rk,
                            const float* qg_k, const float* qg_rk,
                            const float* lg_k, const float* lg_rk,
                            const float* pe_w, const float* ro1_w,
                            const float* ro2_w, _Float16* H) {
  switch (blockIdx.x) {
    case 0:  pack_mat(wt0_rk, H + HV_WT0RK, 32, 96); break;
    case 1:  pack_mat(wt1_rk, H + HV_WT1RK, 32, 96); break;
    case 2:  pack_mat(pg_k,   H + HV_PGK,   64, 96); break;
    case 3:  pack_mat(pg_rk,  H + HV_PGRK,  32, 96); break;
    case 4:  pack_mat(qg_k,   H + HV_QGK,   32, 96); break;
    case 5:  pack_mat(qg_rk,  H + HV_QGRK,  32, 96); break;
    case 6:  pack_mat(lg_k,   H + HV_LGK,   32, 96); break;
    case 7:  pack_mat(lg_rk,  H + HV_LGRK,  32, 96); break;
    case 8:  pack_mat(pe_w,   H + HV_PEW,   64, 32); break;
    case 9:  pack_mat(ro1_w,  H + HV_RO1,   32, 16); break;
    case 10: pack_mat_pad16(ro2_w, H + HV_RO2);      break;
  }
}

// ---------------- link load ----------------
__global__ void load_kernel(const float* ft, const float* cap, const int* p2l, float* loadv) {
  int l = blockIdx.x * blockDim.x + threadIdx.x;
  if (l >= L_N) return;
  float s = 0.f;
  for (int k = 0; k < K_N; ++k) s += ft[p2l[((size_t)l * K_N + k) * 2]];
  loadv[l] = s * fast_rcp(cap[l]);
}

// ---------------- wavelet GRU encoders (2 levels) ----------------
__global__ void __launch_bounds__(128, 1)
wavelet_kernel(const float* ipg1, const float* ps1,
               const float* ipg2, const float* ps2,
               const float* wt0_k, const float* wt0_b,
               const float* wt1_k, const float* wt1_b,
               const _Float16* H, _Float16* enc) {
  __shared__ __align__(16) _Float16 hbuf[4][16 * 32];
  __shared__ float xbuf[4][32];
  int wave = threadIdx.x >> 5, lane = threadIdx.x & 31;
  int tile = blockIdx.x * 4 + wave;
  if (tile * 16 >= P_N) return;
  int p0 = tile * 16;
  int m = lane & 15, hi = lane >> 4;
  int base = hi ? 8 : 0, mb = hi * 8;
  _Float16* hl = hbuf[wave];
  float* xb = xbuf[wave];
  v8f zc = splat8(0.f);

  for (int lvl = 0; lvl < 2; ++lvl) {
    const float* xi = lvl ? ipg2 : ipg1;
    const float* xs = lvl ? ps2 : ps1;
    const float* kk = lvl ? wt1_k : wt0_k;   // (2,96)
    const float* bb = lvl ? wt1_b : wt0_b;   // (2,96)
    const _Float16* rkp = H + (lvl ? HV_WT1RK : HV_WT0RK);

    // --- hoisted invariants: weight fragments + per-lane gate coefficients ---
    v16h rkf[6];
#pragma unroll
    for (int nt = 0; nt < 6; ++nt) rkf[nt] = load_bfrag(rkp, nt);
    float kz0[2], kz1[2], bz[2], kr0[2], kr1[2], br[2], kc0[2], kc1[2], bc0[2], bc1[2];
#pragma unroll
    for (int jt = 0; jt < 2; ++jt) {
      int cz = jt * 16 + m, cr = cz + 32, cc = cz + 64;
      kz0[jt] = kk[cz]; kz1[jt] = kk[96 + cz]; bz[jt] = bb[cz] + bb[96 + cz];
      kr0[jt] = kk[cr]; kr1[jt] = kk[96 + cr]; br[jt] = bb[cr] + bb[96 + cr];
      kc0[jt] = kk[cc]; kc1[jt] = kk[96 + cc]; bc0[jt] = bb[cc]; bc1[jt] = bb[96 + cc];
    }

    v8f h0 = splat8(0.f), h1 = splat8(0.f);
    for (int j = 0; j < 16; ++j) hl[lane * 16 + j] = (_Float16)0.f;
    WAVE_LDS_FENCE();

    for (int t = 0; t < S_N; ++t) {
      if (hi == 0) xb[m] = xi[(size_t)(p0 + m) * S_N + t];
      else         xb[16 + m] = xs[(size_t)(p0 + m) * S_N + t];
      WAVE_LDS_FENCE();

      v16h ah = make_a(hl + m * 32, base);
      v8f acc[6];
#pragma unroll
      for (int nt = 0; nt < 6; ++nt) acc[nt] = wmma16(ah, rkf[nt], zc);
#pragma unroll
      for (int jt = 0; jt < 2; ++jt) {
        v8f hc = jt ? h1 : h0, out;
#pragma unroll
        for (int v = 0; v < 8; ++v) {
          int mr = mb + v;
          float x0 = xb[mr], x1 = xb[16 + mr];
          float z = sigm(acc[jt][v] + x0 * kz0[jt] + x1 * kz1[jt] + bz[jt]);
          float r = sigm(acc[2 + jt][v] + x0 * kr0[jt] + x1 * kr1[jt] + br[jt]);
          float c = tanh_f(x0 * kc0[jt] + x1 * kc1[jt] + bc0[jt] + r * (acc[4 + jt][v] + bc1[jt]));
          out[v] = z * hc[v] + (1.f - z) * c;
        }
        if (jt) h1 = out; else h0 = out;
      }
#pragma unroll
      for (int v = 0; v < 8; ++v) {
        hl[(mb + v) * 32 + m]      = (_Float16)h0[v];
        hl[(mb + v) * 32 + 16 + m] = (_Float16)h1[v];
      }
      WAVE_LDS_FENCE();
    }
#pragma unroll
    for (int v = 0; v < 8; ++v) {
      enc[(size_t)(p0 + mb + v) * 64 + lvl * 32 + m]      = (_Float16)h0[v];
      enc[(size_t)(p0 + mb + v) * 64 + lvl * 32 + 16 + m] = (_Float16)h1[v];
    }
  }
}

// ---------------- path embedding: relu(enc @ W + b) ----------------
__global__ void __launch_bounds__(128, 1)
path_emb_kernel(const _Float16* H, const float* pe_b, float* path_state) {
  int wave = threadIdx.x >> 5, lane = threadIdx.x & 31;
  int tile = blockIdx.x * 4 + wave;
  if (tile * 16 >= P_N) return;
  int p0 = tile * 16;
  int m = lane & 15, hi = lane >> 4;
  int base = hi ? 8 : 0, mb = hi * 8;
  const _Float16* enc = H + HV_ENC;
  const _Float16* pw  = H + HV_PEW;
  v8f zc = splat8(0.f);
  v16h a0 = make_a(enc + (size_t)(p0 + m) * 64, base);
  v16h a1 = make_a(enc + (size_t)(p0 + m) * 64 + 32, base);
#pragma unroll
  for (int nt = 0; nt < 2; ++nt) {
    float bv = pe_b[nt * 16 + m];
    v8f c = wmma16(a0, load_bfrag(pw, nt * 2 + 0), zc);
    c = wmma16(a1, load_bfrag(pw, nt * 2 + 1), c);
#pragma unroll
    for (int v = 0; v < 8; ++v)
      path_state[(size_t)(p0 + mb + v) * 32 + nt * 16 + m] = fmaxf(c[v] + bv, 0.f);
  }
}

// ---------------- link / queue embeddings (VALU, tiny) ----------------
__global__ void link_emb_kernel(const float* loadv, const float* w1, const float* b1,
                                const float* w2, const float* b2,
                                float* ls, _Float16* lf16) {
  int idx = blockIdx.x * blockDim.x + threadIdx.x;
  if (idx >= L_N * 32) return;
  int l = idx >> 5, j = idx & 31;
  float x = loadv[l];
  float acc = b2[j];
  for (int i = 0; i < 32; ++i) {
    float t1 = fmaxf(x * w1[i] + b1[i], 0.f);
    acc += t1 * w2[i * 32 + j];
  }
  acc = fmaxf(acc, 0.f);
  ls[idx] = acc; lf16[idx] = (_Float16)acc;
}

__global__ void queue_emb_kernel(const int* bt, const float* w1, const float* b1,
                                 const float* w2, const float* b2,
                                 float* qs, _Float16* qf16) {
  int idx = blockIdx.x * blockDim.x + threadIdx.x;
  if (idx >= Q_N * 32) return;
  int q = idx >> 5, j = idx & 31;
  int b = bt[q];
  float acc = b2[j];
  for (int i = 0; i < 32; ++i) {
    float t1 = fmaxf(w1[b * 32 + i] + b1[i], 0.f);
    acc += t1 * w2[i * 32 + j];
  }
  acc = fmaxf(acc, 0.f);
  qs[idx] = acc; qf16[idx] = (_Float16)acc;
}

// ---------------- path GRU (8 steps, input 64 = qg||lg) ----------------
__global__ void __launch_bounds__(128, 1)
path_gru_kernel(const _Float16* H, const int* l2p,
                const float* pg_b, float* path_state, float* pss) {
  __shared__ __align__(16) _Float16 hbuf[4][16 * 32];
  int wave = threadIdx.x >> 5, lane = threadIdx.x & 31;
  int tile = blockIdx.x * 4 + wave;
  if (tile * 16 >= P_N) return;
  int p0 = tile * 16;
  int m = lane & 15, hi = lane >> 4;
  int base = hi ? 8 : 0, mb = hi * 8;
  _Float16* hl = hbuf[wave];
  const _Float16* qf = H + HV_QF16;
  const _Float16* lf = H + HV_LF16;
  v8f zc = splat8(0.f);

  // --- hoisted invariants: 12 k-frags, 6 rk-frags, per-lane biases ---
  v16h kf[12], rf[6];
#pragma unroll
  for (int f = 0; f < 12; ++f) kf[f] = load_bfrag(H + HV_PGK, f);
#pragma unroll
  for (int f = 0; f < 6; ++f) rf[f] = load_bfrag(H + HV_PGRK, f);
  float bz[2], br[2], bcx[2], bch[2];
#pragma unroll
  for (int jt = 0; jt < 2; ++jt) {
    int cz = jt * 16 + m;
    bz[jt]  = pg_b[cz] + pg_b[96 + cz];
    br[jt]  = pg_b[cz + 32] + pg_b[96 + cz + 32];
    bcx[jt] = pg_b[cz + 64];
    bch[jt] = pg_b[96 + cz + 64];
  }

  v8f h0, h1;
#pragma unroll
  for (int v = 0; v < 8; ++v) {
    h0[v] = path_state[(size_t)(p0 + mb + v) * 32 + m];
    h1[v] = path_state[(size_t)(p0 + mb + v) * 32 + 16 + m];
  }
#pragma unroll
  for (int v = 0; v < 8; ++v) {
    hl[(mb + v) * 32 + m]      = (_Float16)h0[v];
    hl[(mb + v) * 32 + 16 + m] = (_Float16)h1[v];
    pss[((size_t)(p0 + mb + v) * 9 + 0) * 32 + m]      = h0[v];
    pss[((size_t)(p0 + mb + v) * 9 + 0) * 32 + 16 + m] = h1[v];
  }
  WAVE_LDS_FENCE();

  for (int t = 0; t < T_N; ++t) {
    int idx = l2p[(size_t)(p0 + m) * T_N + t];
    v16h a0 = make_a(qf + (size_t)idx * 32, base);
    v16h a1 = make_a(lf + (size_t)idx * 32, base);
    v16h ah = make_a(hl + m * 32, base);
#pragma unroll
    for (int jt = 0; jt < 2; ++jt) {
      v8f az  = wmma16(ah, rf[jt],
                wmma16(a1, kf[jt * 2 + 1], wmma16(a0, kf[jt * 2 + 0], zc)));
      v8f ar  = wmma16(ah, rf[2 + jt],
                wmma16(a1, kf[(2 + jt) * 2 + 1], wmma16(a0, kf[(2 + jt) * 2 + 0], zc)));
      v8f acx = wmma16(a1, kf[(4 + jt) * 2 + 1], wmma16(a0, kf[(4 + jt) * 2 + 0], zc));
      v8f ach = wmma16(ah, rf[4 + jt], zc);
      v8f hc = jt ? h1 : h0, out;
#pragma unroll
      for (int v = 0; v < 8; ++v) {
        float z = sigm(az[v] + bz[jt]);
        float r = sigm(ar[v] + br[jt]);
        float c = tanh_f(acx[v] + bcx[jt] + r * (ach[v] + bch[jt]));
        out[v] = z * hc[v] + (1.f - z) * c;
      }
      if (jt) h1 = out; else h0 = out;
    }
#pragma unroll
    for (int v = 0; v < 8; ++v) {
      hl[(mb + v) * 32 + m]      = (_Float16)h0[v];
      hl[(mb + v) * 32 + 16 + m] = (_Float16)h1[v];
      pss[((size_t)(p0 + mb + v) * 9 + t + 1) * 32 + m]      = h0[v];
      pss[((size_t)(p0 + mb + v) * 9 + t + 1) * 32 + 16 + m] = h1[v];
    }
    WAVE_LDS_FENCE();
  }
#pragma unroll
  for (int v = 0; v < 8; ++v) {
    path_state[(size_t)(p0 + mb + v) * 32 + m]      = h0[v];
    path_state[(size_t)(p0 + mb + v) * 32 + 16 + m] = h1[v];
  }
}

// ---------------- queue update: gru_cell(sum pg, queue_state) ----------------
__global__ void __launch_bounds__(128, 1)
queue_upd_kernel(_Float16* H, const int* p2l, const float* pss,
                 const float* qg_b, float* qs) {
  __shared__ __align__(16) _Float16 xbuf[4][16 * 32];
  int wave = threadIdx.x >> 5, lane = threadIdx.x & 31;
  int tile = blockIdx.x * 4 + wave;
  if (tile * 16 >= Q_N) return;
  int q0 = tile * 16;
  int m = lane & 15, hi = lane >> 4;
  int base = hi ? 8 : 0, mb = hi * 8;
  _Float16* xl = xbuf[wave];
  _Float16* qf = H + HV_QF16;
  v8f zc = splat8(0.f);

  v16h kfr[6], rfr[6];
#pragma unroll
  for (int f = 0; f < 6; ++f) kfr[f] = load_bfrag(H + HV_QGK, f);
#pragma unroll
  for (int f = 0; f < 6; ++f) rfr[f] = load_bfrag(H + HV_QGRK, f);
  float bz[2], br[2], bcx[2], bch[2];
#pragma unroll
  for (int jt = 0; jt < 2; ++jt) {
    int cz = jt * 16 + m;
    bz[jt]  = qg_b[cz] + qg_b[96 + cz];
    br[jt]  = qg_b[cz + 32] + qg_b[96 + cz + 32];
    bcx[jt] = qg_b[cz + 64];
    bch[jt] = qg_b[96 + cz + 64];
  }

  // sum of gathered path_state_sequence rows; 2 lanes per row (16 dims each)
  float s[16];
#pragma unroll
  for (int d = 0; d < 16; ++d) s[d] = 0.f;
  int qrow = q0 + m;
  for (int k = 0; k < K_N; ++k) {
    int p  = p2l[((size_t)qrow * K_N + k) * 2 + 0];
    int ti = p2l[((size_t)qrow * K_N + k) * 2 + 1];
    const float4* src4 = (const float4*)(pss + ((size_t)p * 9 + ti) * 32 + hi * 16);
#pragma unroll
    for (int j = 0; j < 4; ++j) {
      float4 v4 = src4[j];
      s[4 * j + 0] += v4.x; s[4 * j + 1] += v4.y;
      s[4 * j + 2] += v4.z; s[4 * j + 3] += v4.w;
    }
  }
#pragma unroll
  for (int d = 0; d < 16; ++d) xl[m * 32 + hi * 16 + d] = (_Float16)s[d];
  WAVE_LDS_FENCE();

  v16h ax = make_a(xl + m * 32, base);
  v16h ah = make_a(qf + (size_t)(q0 + m) * 32, base);
  v8f h0, h1;
#pragma unroll
  for (int v = 0; v < 8; ++v) {
    h0[v] = qs[(size_t)(q0 + mb + v) * 32 + m];
    h1[v] = qs[(size_t)(q0 + mb + v) * 32 + 16 + m];
  }
#pragma unroll
  for (int jt = 0; jt < 2; ++jt) {
    v8f az  = wmma16(ah, rfr[jt],     wmma16(ax, kfr[jt], zc));
    v8f ar  = wmma16(ah, rfr[2 + jt], wmma16(ax, kfr[2 + jt], zc));
    v8f acx = wmma16(ax, kfr[4 + jt], zc);
    v8f ach = wmma16(ah, rfr[4 + jt], zc);
    v8f hc = jt ? h1 : h0;
#pragma unroll
    for (int v = 0; v < 8; ++v) {
      float z = sigm(az[v] + bz[jt]);
      float r = sigm(ar[v] + br[jt]);
      float c = tanh_f(acx[v] + bcx[jt] + r * (ach[v] + bch[jt]));
      float hn = z * hc[v] + (1.f - z) * c;
      size_t o = (size_t)(q0 + mb + v) * 32 + jt * 16 + m;
      qs[o] = hn; qf[o] = (_Float16)hn;
    }
  }
}

// ---------------- link update: 2-step GRU over gathered queue states ----------------
__global__ void __launch_bounds__(128, 1)
link_upd_kernel(_Float16* H, const int* q2l, const float* lg_b, float* ls) {
  __shared__ __align__(16) _Float16 hbuf[4][16 * 32];
  int wave = threadIdx.x >> 5, lane = threadIdx.x & 31;
  int tile = blockIdx.x * 4 + wave;
  if (tile * 16 >= L_N) return;
  int l0 = tile * 16;
  int m = lane & 15, hi = lane >> 4;
  int base = hi ? 8 : 0, mb = hi * 8;
  _Float16* hl = hbuf[wave];
  const _Float16* qf = H + HV_QF16;
  _Float16* lf = H + HV_LF16;
  v8f zc = splat8(0.f);

  v16h kfr[6], rfr[6];
#pragma unroll
  for (int f = 0; f < 6; ++f) kfr[f] = load_bfrag(H + HV_LGK, f);
#pragma unroll
  for (int f = 0; f < 6; ++f) rfr[f] = load_bfrag(H + HV_LGRK, f);
  float bz[2], br[2], bcx[2], bch[2];
#pragma unroll
  for (int jt = 0; jt < 2; ++jt) {
    int cz = jt * 16 + m;
    bz[jt]  = lg_b[cz] + lg_b[96 + cz];
    br[jt]  = lg_b[cz + 32] + lg_b[96 + cz + 32];
    bcx[jt] = lg_b[cz + 64];
    bch[jt] = lg_b[96 + cz + 64];
  }

  v8f h0, h1;
#pragma unroll
  for (int v = 0; v < 8; ++v) {
    h0[v] = ls[(size_t)(l0 + mb + v) * 32 + m];
    h1[v] = ls[(size_t)(l0 + mb + v) * 32 + 16 + m];
  }
#pragma unroll
  for (int v = 0; v < 8; ++v) {
    hl[(mb + v) * 32 + m]      = (_Float16)h0[v];
    hl[(mb + v) * 32 + 16 + m] = (_Float16)h1[v];
  }
  WAVE_LDS_FENCE();

  for (int t = 0; t < 2; ++t) {
    int idx = q2l[(size_t)(l0 + m) * 2 + t];
    v16h ax = make_a(qf + (size_t)idx * 32, base);
    v16h ah = make_a(hl + m * 32, base);
#pragma unroll
    for (int jt = 0; jt < 2; ++jt) {
      v8f az  = wmma16(ah, rfr[jt],     wmma16(ax, kfr[jt], zc));
      v8f ar  = wmma16(ah, rfr[2 + jt], wmma16(ax, kfr[2 + jt], zc));
      v8f acx = wmma16(ax, kfr[4 + jt], zc);
      v8f ach = wmma16(ah, rfr[4 + jt], zc);
      v8f hc = jt ? h1 : h0, out;
#pragma unroll
      for (int v = 0; v < 8; ++v) {
        float z = sigm(az[v] + bz[jt]);
        float r = sigm(ar[v] + br[jt]);
        float c = tanh_f(acx[v] + bcx[jt] + r * (ach[v] + bch[jt]));
        out[v] = z * hc[v] + (1.f - z) * c;
      }
      if (jt) h1 = out; else h0 = out;
    }
#pragma unroll
    for (int v = 0; v < 8; ++v) {
      hl[(mb + v) * 32 + m]      = (_Float16)h0[v];
      hl[(mb + v) * 32 + 16 + m] = (_Float16)h1[v];
    }
    WAVE_LDS_FENCE();
  }
#pragma unroll
  for (int v = 0; v < 8; ++v) {
    size_t o0 = (size_t)(l0 + mb + v) * 32 + m;
    size_t o1 = o0 + 16;
    ls[o0] = h0[v]; ls[o1] = h1[v];
    lf[o0] = (_Float16)h0[v]; lf[o1] = (_Float16)h1[v];
  }
}

// ---------------- readout: WMMA MLP over flattened (path, t) rows ----------------
// rows r = p*8 + t; 16-row tiles => 2 paths per tile (rows 0..7 -> path A, 8..15 -> path B)
__global__ void __launch_bounds__(128, 1)
readout_kernel(const float* pss, const float* ft, const float* fp,
               const float* cap, const int* l2p, const _Float16* H,
               const float* b1, const float* b2, const float* b3,
               const float* w3, float* out) {
  __shared__ __align__(16) _Float16 hbuf[4][16 * 32];
  int wave = threadIdx.x >> 5, lane = threadIdx.x & 31;
  int tile = blockIdx.x * 4 + wave;
  if (tile * 16 >= P_N * T_N) return;
  int r0 = tile * 16;
  int m = lane & 15, hi = lane >> 4;
  int base = hi ? 8 : 0, mb = hi * 8;
  _Float16* hl = hbuf[wave];
  v8f zc = splat8(0.f);

  // zero LDS tile (upper 16 cols stay zero => K padding for layer 2)
  for (int j = 0; j < 16; ++j) hl[lane * 16 + j] = (_Float16)0.f;

  // layer 1 A-fragment straight from fp32 pss with on-the-fly f16 convert
  int r = r0 + m;
  int pr = r >> 3, tr = r & 7;
  const float* src = pss + ((size_t)pr * 9 + tr + 1) * 32;
  v16h a1;
#pragma unroll
  for (int i = 0; i < 8; ++i) a1[i] = (_Float16)src[base + i];
#pragma unroll
  for (int i = 0; i < 8; ++i) a1[8 + i] = (_Float16)src[base + 16 + i];

  v8f c1 = wmma16(a1, load_bfrag(H + HV_RO1, 0), zc);
  float bv1 = b1[m];
  WAVE_LDS_FENCE();  // zero-fill visible before overwriting cols 0..15
#pragma unroll
  for (int v = 0; v < 8; ++v)
    hl[(mb + v) * 32 + m] = (_Float16)fmaxf(c1[v] + bv1, 0.f);
  WAVE_LDS_FENCE();

  v16h a2 = make_a(hl + m * 32, base);
  v8f c2 = wmma16(a2, load_bfrag(H + HV_RO2, 0), zc);
  float bv2 = b2[m];
  float w3n = w3[m];
  float part[8];
#pragma unroll
  for (int v = 0; v < 8; ++v) part[v] = fmaxf(c2[v] + bv2, 0.f) * w3n;
  // reduce across the 16 lanes of each half (columns) -> occ per row
#pragma unroll
  for (int off = 1; off < 16; off <<= 1) {
#pragma unroll
    for (int v = 0; v < 8; ++v) part[v] += __shfl_xor(part[v], off, 32);
  }
  // rows mb..mb+15? this half covers rows mb..mb+7 == one path
  int pp = (r0 + mb) >> 3;  // path for this half-tile
  float b3v = b3[0];
  float qd = 0.f, invs = 0.f;
#pragma unroll
  for (int v = 0; v < 8; ++v) {
    float icg = fast_rcp(cap[l2p[(size_t)pp * T_N + v]] * 1e9f);
    qd += (part[v] + b3v) * icg;
    invs += icg;
  }
  if (m == 0) out[pp] = qd + (ft[pp] * fast_rcp(fp[pp])) * invs;
}

// ---------------- host launcher ----------------
extern "C" void kernel_launch(void* const* d_in, const int* in_sizes, int n_in,
                              void* d_out, int out_size, void* d_ws, size_t ws_size,
                              hipStream_t stream) {
  (void)in_sizes; (void)n_in; (void)out_size; (void)ws_size;
  // array inputs (setup_inputs order)
  const float* flow_traffic  = (const float*)d_in[0];
  const float* flow_packets  = (const float*)d_in[1];
  const float* link_capacity = (const float*)d_in[3];
  const int*   buffer_type   = (const int*)d_in[4];
  const int*   link_to_path  = (const int*)d_in[5];
  const int*   path_to_link  = (const int*)d_in[6];
  const int*   queue_to_link = (const int*)d_in[7];
  const float* ipg1 = (const float*)d_in[8];
  const float* ps1  = (const float*)d_in[9];
  const float* ipg2 = (const float*)d_in[10];
  const float* ps2  = (const float*)d_in[11];
  // params pytree, sorted-key order
  const float* le1_b = (const float*)d_in[12];
  const float* le1_w = (const float*)d_in[13];
  const float* le2_b = (const float*)d_in[14];
  const float* le2_w = (const float*)d_in[15];
  const float* lg_b  = (const float*)d_in[16];
  const float* lg_k  = (const float*)d_in[17];
  const float* lg_rk = (const float*)d_in[18];
  const float* pe_b  = (const float*)d_in[19];
  const float* pe_w  = (const float*)d_in[20];
  const float* pg_b  = (const float*)d_in[21];
  const float* pg_k  = (const float*)d_in[22];
  const float* pg_rk = (const float*)d_in[23];
  const float* qe1_b = (const float*)d_in[24];
  const float* qe1_w = (const float*)d_in[25];
  const float* qe2_b = (const float*)d_in[26];
  const float* qe2_w = (const float*)d_in[27];
  const float* qg_b  = (const float*)d_in[28];
  const float* qg_k  = (const float*)d_in[29];
  const float* qg_rk = (const float*)d_in[30];
  const float* ro1_b = (const float*)d_in[31];
  const float* ro1_w = (const float*)d_in[32];
  const float* ro2_b = (const float*)d_in[33];
  const float* ro2_w = (const float*)d_in[34];
  const float* ro3_b = (const float*)d_in[35];
  const float* ro3_w = (const float*)d_in[36];
  const float* wt0_b  = (const float*)d_in[37];
  const float* wt0_k  = (const float*)d_in[38];
  const float* wt0_rk = (const float*)d_in[39];
  const float* wt1_b  = (const float*)d_in[40];
  const float* wt1_k  = (const float*)d_in[41];
  const float* wt1_rk = (const float*)d_in[42];

  _Float16* H = (_Float16*)d_ws;
  float* F = (float*)((char*)d_ws + F32_BASE_BYTES);
  float* path_state = F + FO_PS;
  float* pss        = F + FO_PSS;
  float* qs         = F + FO_QS;
  float* ls         = F + FO_LS;
  float* loadv      = F + FO_LOAD;
  float* out        = (float*)d_out;

  // 1. pack weights into WMMA B-fragment layout
  pack_kernel<<<11, 256, 0, stream>>>(wt0_rk, wt1_rk, pg_k, pg_rk, qg_k, qg_rk,
                                      lg_k, lg_rk, pe_w, ro1_w, ro2_w, H);
  // 2. link load
  load_kernel<<<(L_N + 255) / 256, 256, 0, stream>>>(flow_traffic, link_capacity,
                                                     path_to_link, loadv);
  // 3. wavelet encoders
  wavelet_kernel<<<(P_N / 16 + 3) / 4, 128, 0, stream>>>(
      ipg1, ps1, ipg2, ps2, wt0_k, wt0_b, wt1_k, wt1_b, H, H + HV_ENC);
  // 4. embeddings
  path_emb_kernel<<<(P_N / 16 + 3) / 4, 128, 0, stream>>>(H, pe_b, path_state);
  link_emb_kernel<<<(L_N * 32 + 255) / 256, 256, 0, stream>>>(
      loadv, le1_w, le1_b, le2_w, le2_b, ls, H + HV_LF16);
  queue_emb_kernel<<<(Q_N * 32 + 255) / 256, 256, 0, stream>>>(
      buffer_type, qe1_w, qe1_b, qe2_w, qe2_b, qs, H + HV_QF16);
  // 5. message passing
  for (int it = 0; it < 8; ++it) {
    path_gru_kernel<<<(P_N / 16 + 3) / 4, 128, 0, stream>>>(
        H, link_to_path, pg_b, path_state, pss);
    queue_upd_kernel<<<(Q_N / 16 + 3) / 4, 128, 0, stream>>>(
        H, path_to_link, pss, qg_b, qs);
    link_upd_kernel<<<(L_N / 16 + 3) / 4, 128, 0, stream>>>(
        H, queue_to_link, lg_b, ls);
  }
  // 6. readout (WMMA MLP)
  readout_kernel<<<(P_N * T_N / 16 + 3) / 4, 128, 0, stream>>>(
      pss, flow_traffic, flow_packets, link_capacity, link_to_path, H,
      ro1_b, ro2_b, ro3_b, ro3_w, out);
}
